// SCTDN_68590627717380
// MI455X (gfx1250) — matrix operand order repaired and checked by reading, not compiled, not run on hardware
//
#include <hip/hip_runtime.h>
#include <hip/hip_bf16.h>

typedef _Float16 h16;
typedef __attribute__((ext_vector_type(16))) _Float16 v16h;
typedef __attribute__((ext_vector_type(8)))  float    v8f;

#define BATCH   16
#define NPTS    4096
#define TSTEPS  30

static __host__ __device__ inline int cdiv(int a, int b) { return (a + b - 1) / b; }

// Weight pre-conversion: f32 -> f16 (one tiny launch per layer, stream-ordered)
__global__ void k_cvtw(const float* __restrict__ in, h16* __restrict__ out, int n)
{
    int t = blockIdx.x * blockDim.x + threadIdx.x;
    if (t < n) out[t] = (h16)in[t];
}

// ---------------------------------------------------------------------------
// Fused GEMM:  Y[M,N] = act( (A[M,K] @ W[N,K]^T) * (g/sqrt(1+1e-5)) + bias )
// act: 0 = identity, 1 = relu, 2 = leaky_relu(0.01)
// wave32-native: each wave computes a 16x64 slab (four adjacent 16x16 N-tiles)
// sharing one A fragment -> A streamed once per 64 output columns.  K consumed
// 32/issue via v_wmma_f32_16x16x32_f16; A (f16) and W (pre-cvt f16) fragments
// are contiguous 16B runs (global_load_b128, zero cvt in the hot loop).  The
// numN==4 interior case is unswitched into a branch-free body.  8 waves per
// block stack along M and share the same W slice (hot in WGP$/L2).
// ---------------------------------------------------------------------------
template <typename TA, typename TY>
__global__ void k_gemm_wmma(const TA* __restrict__ A, const h16* __restrict__ Wt,
                            const float* __restrict__ g, const float* __restrict__ bias,
                            TY* __restrict__ Y, int M, int N, int K, int act)
{
    const int lane    = threadIdx.x & 31;
    const int wave    = threadIdx.x >> 5;
    const int tilesN  = (N + 15) >> 4;
    const int tilesM  = (M + 15) >> 4;
    const int tilesNQ = (tilesN + 3) >> 2;
    const int tnq = blockIdx.x % tilesNQ;
    const int tm  = (blockIdx.x / tilesNQ) * 8 + wave;
    if (tm >= tilesM) return;               // wave-uniform: EXEC stays all-1s

    const int tnb  = tnq * 4;
    const int numN = min(4, tilesN - tnb);  // 1..4, wave-uniform

    const int hrow  = lane & 15;
    const int hi    = lane >> 4;            // half-wave select
    const int aKoff = hi ? 16 : 0;          // contiguous k order (k = 16h + s), matches B
    const int bKoff = hi ? 16 : 0;          // contiguous k order (k = 16h + s), matches A
    const int ar    = tm * 16 + hrow;       // A row this lane streams
    const bool full = (tm * 16 + 16 <= M) && ((tnb + numN) * 16 <= N);
    const int  K32  = K & ~31;

    v8f acc[4] = {{}, {}, {}, {}};

    if (full) {
        const TA* __restrict__ Ap = A + (size_t)ar * K;
        const h16* Wp[4];
#pragma unroll
        for (int q = 0; q < 4; ++q) {
            int tq = tnb + ((q < numN) ? q : 0);
            Wp[q] = Wt + (size_t)(tq * 16 + hrow) * K;
        }
        if (numN == 4) {
            // ---- branch-free interior body (the dominant case) ----
            for (int k0 = 0; k0 < K32; k0 += 32) {
                if (k0 + 32 < K32)
                    __builtin_prefetch(Ap + k0 + 32, 0, 1);
                v16h a;
                const TA* a1 = Ap + k0 + aKoff;       // 8 contiguous elements
                const TA* a2 = a1 + 8;                // 8 contiguous elements
#pragma unroll
                for (int j = 0; j < 8; ++j) {
                    a[j]     = (_Float16)a1[j];
                    a[8 + j] = (_Float16)a2[j];
                }
                v16h b[4];
#pragma unroll
                for (int q = 0; q < 4; ++q) {
                    const h16* bp = Wp[q] + k0 + bKoff;   // 16 contiguous halfs
#pragma unroll
                    for (int j = 0; j < 16; ++j) b[q][j] = bp[j];
                }
#pragma unroll
                for (int q = 0; q < 4; ++q)
                    acc[q] = __builtin_amdgcn_wmma_f32_16x16x32_f16(false, a, false, b[q],
                                                                    (short)0, acc[q], false, false);
            }
        } else {
            for (int k0 = 0; k0 < K32; k0 += 32) {
                if (k0 + 32 < K32)
                    __builtin_prefetch(Ap + k0 + 32, 0, 1);
                v16h a;
                const TA* a1 = Ap + k0 + aKoff;
                const TA* a2 = a1 + 8;
#pragma unroll
                for (int j = 0; j < 8; ++j) {
                    a[j]     = (_Float16)a1[j];
                    a[8 + j] = (_Float16)a2[j];
                }
                v16h b[4];
#pragma unroll
                for (int q = 0; q < 4; ++q) {
                    if (q >= numN) break;
                    const h16* bp = Wp[q] + k0 + bKoff;
#pragma unroll
                    for (int j = 0; j < 16; ++j) b[q][j] = bp[j];
                }
#pragma unroll
                for (int q = 0; q < 4; ++q) {
                    if (q >= numN) break;
                    acc[q] = __builtin_amdgcn_wmma_f32_16x16x32_f16(false, a, false, b[q],
                                                                    (short)0, acc[q], false, false);
                }
            }
        }
    }

    // Guarded path: whole K for edge tiles, or the ragged K tail for interior
    // tiles.  Clamped loads + branch-free selects (no exec-mask churn).
    const int kstart = full ? K32 : 0;
    if (kstart < K) {
        const int arc = (ar < M) ? ar : (M - 1);
        const TA* __restrict__ Ap = A + (size_t)arc * K;
        for (int k0 = kstart; k0 < K; k0 += 32) {
            v16h a;
#pragma unroll
            for (int j = 0; j < 8; ++j) {
                int k1 = k0 + aKoff + j;
                int k2 = k0 + aKoff + 8 + j;
                int k1c = (k1 < K) ? k1 : (K - 1);
                int k2c = (k2 < K) ? k2 : (K - 1);
                float va1 = (float)Ap[k1c];
                float va2 = (float)Ap[k2c];
                a[j]     = (ar < M && k1 < K) ? (_Float16)va1 : (_Float16)0.0f;
                a[8 + j] = (ar < M && k2 < K) ? (_Float16)va2 : (_Float16)0.0f;
            }
#pragma unroll
            for (int q = 0; q < 4; ++q) {
                if (q >= numN) break;
                const int wr  = (tnb + q) * 16 + hrow;
                const int wrc = (wr < N) ? wr : (N - 1);
                const h16* Wp = Wt + (size_t)wrc * K;
                v16h b;
#pragma unroll
                for (int j = 0; j < 16; ++j) {
                    int kk  = k0 + bKoff + j;
                    int kkc = (kk < K) ? kk : (K - 1);
                    h16 vb = Wp[kkc];
                    b[j] = (wr < N && kk < K) ? vb : (h16)0.0f;
                }
                acc[q] = __builtin_amdgcn_wmma_f32_16x16x32_f16(false, a, false, b,
                                                                (short)0, acc[q], false, false);
            }
        }
    }

    // Epilogue: f32 C/D layout -> lane (col, 8 rows), fused BN + activation.
    const int rbase = tm * 16 + (hi ? 8 : 0);
#pragma unroll
    for (int q = 0; q < 4; ++q) {
        if (q >= numN) break;
        const int oc = (tnb + q) * 16 + (lane & 15);
        if (oc < N) {
            float scale = g    ? g[oc] * 0.99999500003749969f : 1.0f;  // 1/sqrt(1+1e-5)
            float bb    = bias ? bias[oc] : 0.0f;
#pragma unroll
            for (int r = 0; r < 8; ++r) {
                int orow = rbase + r;
                if (orow < M) {
                    float y = acc[q][r] * scale + bb;
                    if (act == 1)      y = fmaxf(y, 0.0f);
                    else if (act == 2) y = (y > 0.0f) ? y : 0.01f * y;
                    Y[(size_t)orow * N + oc] = (TY)y;
                }
            }
        }
    }
}

// ---------------------------------------------------------------------------
// Farthest point sampling: one block per batch, dist[] held in LDS.
// ---------------------------------------------------------------------------
__global__ void k_fps(const float* __restrict__ xyz, int n, int npoint, int* __restrict__ out)
{
    __shared__ float dist[4096];
    __shared__ float rv[256];
    __shared__ int   ri[256];
    __shared__ int   lastIdx;
    const int b = blockIdx.x;
    const float* p = xyz + (size_t)b * n * 3;
    const int tid = threadIdx.x;
    for (int i = tid; i < n; i += 256) dist[i] = 1e10f;
    if (tid == 0) lastIdx = 0;
    __syncthreads();
    for (int s = 0; s < npoint; ++s) {
        int lp = lastIdx;
        if (tid == 0) out[b * npoint + s] = lp;
        float lx = p[lp * 3], ly = p[lp * 3 + 1], lz = p[lp * 3 + 2];
        float best = -1.0f; int bi = 0;
        for (int i = tid; i < n; i += 256) {
            float dx = p[i * 3] - lx, dy = p[i * 3 + 1] - ly, dz = p[i * 3 + 2] - lz;
            float d = dx * dx + dy * dy + dz * dz;
            float nd = fminf(dist[i], d);
            dist[i] = nd;
            if (nd > best) { best = nd; bi = i; }
        }
        rv[tid] = best; ri[tid] = bi;
        __syncthreads();
        for (int off = 128; off > 0; off >>= 1) {
            if (tid < off && rv[tid + off] > rv[tid]) { rv[tid] = rv[tid + off]; ri[tid] = ri[tid + off]; }
            __syncthreads();
        }
        if (tid == 0) lastIdx = ri[0];
        __syncthreads();
    }
}

// Ball query: one thread per center; sequential scan gives ascending indices.
__global__ void k_ball(const float* __restrict__ xyz, const float* __restrict__ cent,
                       int n, int m, float r2, int ns, int* __restrict__ out, int bm)
{
    int t = blockIdx.x * blockDim.x + threadIdx.x;
    if (t >= bm) return;
    int b = t / m;
    const float* p = xyz + (size_t)b * n * 3;
    float cx = cent[t * 3], cy = cent[t * 3 + 1], cz = cent[t * 3 + 2];
    int* o = out + (size_t)t * ns;
    int cnt = 0, first = 0; bool found = false;
    for (int i = 0; i < n && cnt < ns; ++i) {
        float dx = p[i * 3] - cx, dy = p[i * 3 + 1] - cy, dz = p[i * 3 + 2] - cz;
        if (dx * dx + dy * dy + dz * dz <= r2) {
            if (!found) { found = true; first = i; }
            o[cnt++] = i;
        }
    }
    for (; cnt < ns; ++cnt) o[cnt] = first;
}

__global__ void k_gather(const float* __restrict__ pts, const int* __restrict__ idx,
                         float* __restrict__ out, int n, int m, int C, int total)
{
    int t = blockIdx.x * blockDim.x + threadIdx.x;
    if (t >= total) return;                  // total = B*m*C
    int c = t % C; int bj = t / C; int b = bj / m;
    out[t] = pts[((size_t)b * n + idx[bj]) * C + c];
}

// grouped tensor: out[b,j,s,:] = [xyz[gidx]-center , feats[gidx]]  (f16 out)
template <typename TF>
__global__ void k_group(const float* __restrict__ xyz, const TF* __restrict__ feats,
                        const float* __restrict__ cent, const int* __restrict__ gidx,
                        h16* __restrict__ out, int n, int m, int ns, int C, int total)
{
    int t = blockIdx.x * blockDim.x + threadIdx.x;
    if (t >= total) return;                  // total = B*m*ns
    int bj = t / ns; int b = bj / m;
    int gi = gidx[t];
    const float* ps = xyz + ((size_t)b * n + gi) * 3;
    const float* ct = cent + (size_t)bj * 3;
    h16* o = out + (size_t)t * (3 + C);
    o[0] = (h16)(ps[0] - ct[0]); o[1] = (h16)(ps[1] - ct[1]); o[2] = (h16)(ps[2] - ct[2]);
    const TF* f = feats + ((size_t)b * n + gi) * C;
    for (int c = 0; c < C; ++c) o[3 + c] = (h16)(float)f[c];
}

__global__ void k_maxpool(const h16* __restrict__ x, h16* __restrict__ out,
                          int ns, int C, int total)
{
    int t = blockIdx.x * blockDim.x + threadIdx.x;
    if (t >= total) return;                  // total = B*m*C
    int c = t % C; int bj = t / C;
    const h16* p = x + ((size_t)bj * ns) * C + c;
    float v = -1e30f;
    for (int s = 0; s < ns; ++s) v = fmaxf(v, (float)p[(size_t)s * C]);
    out[t] = (h16)v;
}

__global__ void k_three_nn(const float* __restrict__ x1, const float* __restrict__ x2,
                           int n1, int n2, int* __restrict__ oid, float* __restrict__ ow, int total)
{
    int t = blockIdx.x * blockDim.x + threadIdx.x;
    if (t >= total) return;                  // total = B*n1
    int b = t / n1;
    float px = x1[t * 3], py = x1[t * 3 + 1], pz = x1[t * 3 + 2];
    const float* q = x2 + (size_t)b * n2 * 3;
    float d0 = 1e30f, d1 = 1e30f, d2 = 1e30f; int i0 = 0, i1 = 0, i2 = 0;
    for (int i = 0; i < n2; ++i) {
        float dx = q[i * 3] - px, dy = q[i * 3 + 1] - py, dz = q[i * 3 + 2] - pz;
        float d = dx * dx + dy * dy + dz * dz;
        if (d < d0)      { d2 = d1; i2 = i1; d1 = d0; i1 = i0; d0 = d; i0 = i; }
        else if (d < d1) { d2 = d1; i2 = i1; d1 = d;  i1 = i; }
        else if (d < d2) { d2 = d;  i2 = i; }
    }
    float w0 = 1.0f / (d0 + 1e-8f), w1 = 1.0f / (d1 + 1e-8f), w2 = 1.0f / (d2 + 1e-8f);
    float ws = w0 + w1 + w2;
    oid[t * 3] = i0; oid[t * 3 + 1] = i1; oid[t * 3 + 2] = i2;
    ow[t * 3] = w0 / ws; ow[t * 3 + 1] = w1 / ws; ow[t * 3 + 2] = w2 / ws;
}

template <typename T1>
__global__ void k_interp_cat(const T1* __restrict__ f1, const h16* __restrict__ f2,
                             const int* __restrict__ idx3, const float* __restrict__ w3,
                             h16* __restrict__ out, int n1, int n2, int C1, int C2, int total)
{
    int t = blockIdx.x * blockDim.x + threadIdx.x;
    if (t >= total) return;                  // total = B*n1*(C1+C2)
    int C = C1 + C2;
    int c = t % C; int bi = t / C; int b = bi / n1;
    if (c < C1) { out[t] = (h16)(float)f1[(size_t)bi * C1 + c]; return; }
    int cc = c - C1;
    const int*   id = idx3 + (size_t)bi * 3;
    const float* w  = w3   + (size_t)bi * 3;
    const h16*   fb = f2 + (size_t)b * n2 * C2;
    float v = w[0] * (float)fb[(size_t)id[0] * C2 + cc] +
              w[1] * (float)fb[(size_t)id[1] * C2 + cc] +
              w[2] * (float)fb[(size_t)id[2] * C2 + cc];
    out[t] = (h16)v;
}

__global__ void k_aff(const h16* __restrict__ whole, const float* __restrict__ w,
                      const float* __restrict__ b, float* __restrict__ out, int total)
{
    int t = blockIdx.x * blockDim.x + threadIdx.x;
    if (t >= total) return;                  // total = B*N
    const h16* x = whole + (size_t)t * 32;
    float s = b[0];
    for (int c = 0; c < 32; ++c) s += (float)x[c] * w[c];
    out[t] = 1.0f / (1.0f + expf(-s));
}

__global__ void k_cat_xyzf(const float* __restrict__ xyz, const h16* __restrict__ f,
                           h16* __restrict__ out, int C, int total)
{
    int t = blockIdx.x * blockDim.x + threadIdx.x;
    if (t >= total) return;                  // total = B*m*(3+C)
    int CC = 3 + C;
    int c = t % CC; int bj = t / CC;
    out[t] = (c < 3) ? (h16)xyz[(size_t)bj * 3 + c] : f[(size_t)bj * C + (c - 3)];
}

__global__ void k_softmax4(const float* __restrict__ in, float* __restrict__ out, int rows)
{
    int t = blockIdx.x * blockDim.x + threadIdx.x;
    if (t >= rows) return;
    float a = in[t * 4], b = in[t * 4 + 1], c = in[t * 4 + 2], d = in[t * 4 + 3];
    float mx = fmaxf(fmaxf(a, b), fmaxf(c, d));
    float ea = expf(a - mx), eb = expf(b - mx), ec = expf(c - mx), ed = expf(d - mx);
    float s = ea + eb + ec + ed;
    out[t * 4] = ea / s; out[t * 4 + 1] = eb / s; out[t * 4 + 2] = ec / s; out[t * 4 + 3] = ed / s;
}

__global__ void k_tt(const float* __restrict__ traj, const float* __restrict__ pcs,
                     h16* __restrict__ out, int total)
{
    int t = blockIdx.x * blockDim.x + threadIdx.x;
    if (t >= total) return;                  // total = B*T*9
    int d = t % 9; int bt = t / 9; int b = bt / TSTEPS;
    float v = traj[t];
    if (d < 3) v -= (traj[(size_t)b * TSTEPS * 9 + d] - pcs[(size_t)b * NPTS * 3 + d]);
    out[t] = (h16)v;
}

__global__ void k_fall(const h16* __restrict__ whole, const h16* __restrict__ ftraj,
                       float* __restrict__ out, int total)
{
    int t = blockIdx.x * blockDim.x + threadIdx.x;
    if (t >= total) return;                  // total = B*T*96
    int c = t % 96; int bt = t / 96; int b = bt / TSTEPS;
    out[t] = (c < 32) ? (float)whole[(size_t)b * NPTS * 32 + c]
                      : (float)ftraj[(size_t)bt * 64 + (c - 32)];
}

__global__ void k_lstm(const float* __restrict__ x, const float* __restrict__ Wih,
                       const float* __restrict__ Whh, const float* __restrict__ bih,
                       const float* __restrict__ bhh, float* __restrict__ hs)
{
    __shared__ float h[128];
    int b = blockIdx.x; int u = threadIdx.x;
    h[u] = 0.0f; float c = 0.0f;
    __syncthreads();
    for (int t = 0; t < TSTEPS; ++t) {
        const float* xt = x + ((size_t)b * TSTEPS + t) * 96;
        float gi = bih[u] + bhh[u];
        float gf = bih[128 + u] + bhh[128 + u];
        float gg = bih[256 + u] + bhh[256 + u];
        float go = bih[384 + u] + bhh[384 + u];
        for (int k = 0; k < 96; ++k) {
            float xv = xt[k];
            gi += xv * Wih[(size_t)u * 96 + k];
            gf += xv * Wih[(size_t)(128 + u) * 96 + k];
            gg += xv * Wih[(size_t)(256 + u) * 96 + k];
            go += xv * Wih[(size_t)(384 + u) * 96 + k];
        }
        for (int k = 0; k < 128; ++k) {
            float hv = h[k];
            gi += hv * Whh[(size_t)u * 128 + k];
            gf += hv * Whh[(size_t)(128 + u) * 128 + k];
            gg += hv * Whh[(size_t)(256 + u) * 128 + k];
            go += hv * Whh[(size_t)(384 + u) * 128 + k];
        }
        float si = 1.0f / (1.0f + expf(-gi));
        float sf = 1.0f / (1.0f + expf(-gf));
        float so = 1.0f / (1.0f + expf(-go));
        c = sf * c + si * tanhf(gg);
        float hn = so * tanhf(c);
        __syncthreads();
        h[u] = hn;
        hs[((size_t)b * TSTEPS + t) * 128 + u] = hn;
        __syncthreads();
    }
}

// ---------------------------------------------------------------------------
// Host orchestration
// ---------------------------------------------------------------------------
template <typename TA, typename TY>
static void gemm(const TA* A, const float* Wf, const float* g, const float* bias,
                 TY* Y, int M, int N, int K, int act, h16* Wh, hipStream_t s)
{
    k_cvtw<<<cdiv(N * K, 256), 256, 0, s>>>(Wf, Wh, N * K);   // stream-ordered
    int tilesN = cdiv(N, 16), tilesM = cdiv(M, 16);
    int tilesNQ = cdiv(tilesN, 4);
    int blocks = tilesNQ * cdiv(tilesM, 8);
    k_gemm_wmma<TA, TY><<<blocks, 256, 0, s>>>(A, Wh, g, bias, Y, M, N, K, act);
}

static const h16* run_mlp(const h16* in, h16* GA, h16* GB, h16* outf,
                          const float* const* Ws, const float* const* Gs,
                          const float* const* Bs, const int* ch, int L, int rows,
                          h16* Wh, hipStream_t s)
{
    const h16* cur = in;
    for (int li = 0; li < L; ++li) {
        h16* dst = (li == L - 1) ? outf : ((cur == GA) ? GB : GA);
        gemm<h16, h16>(cur, Ws[li], Gs[li], Bs[li], dst, rows, ch[li + 1], ch[li], 1, Wh, s);
        cur = dst;
    }
    return cur;
}

extern "C" void kernel_launch(void* const* d_in, const int* in_sizes, int n_in,
                              void* d_out, int out_size, void* d_ws, size_t ws_size,
                              hipStream_t stream)
{
    (void)in_sizes; (void)n_in; (void)out_size; (void)ws_size;
    auto F = [&](int i) { return (const float*)d_in[i]; };
    auto mlpP = [&](int base, int L, const float** Ws, const float** Gs, const float** Bs) {
        for (int i = 0; i < L; ++i) {
            Ws[i] = F(base + 3 * i); Gs[i] = F(base + 3 * i + 1); Bs[i] = F(base + 3 * i + 2);
        }
    };
    const float* pcs  = F(0);   // (16,4096,3)
    const float* traj = F(1);   // (16,30,9)
    float* dout = (float*)d_out;

    // ---- workspace arena (allocated in float units; some buffers cast to h16) ----
    float* W0 = (float*)d_ws; size_t off = 0;
    auto alloc = [&](size_t n) { float* p = W0 + off; off += n; return p; };
    h16* GA = (h16*)alloc(16777216);  h16* GB = (h16*)alloc(16777216);   // 33.5M h16 each
    h16* Wh = (h16*)alloc(98304);                                        // max 196608 h16 weights
    int* fidx = (int*)alloc(8192); int* gidx = (int*)alloc(262144);
    int* nnid = (int*)alloc(196608); float* nnw = alloc(196608);
    float* l1x = alloc(24576); float* l2x = alloc(12288); float* l3x = alloc(3072); float* l4x = alloc(768);
    h16* l1f = (h16*)alloc(262144); h16* l2f = (h16*)alloc(262144);
    h16* l3f = (h16*)alloc(131072); h16* l4f = (h16*)alloc(65536);
    h16* n3f = (h16*)alloc(131072); h16* n2f = (h16*)alloc(524288);
    h16* n1f = (h16*)alloc(524288); h16* l0f = (h16*)alloc(4194304);
    h16* whole = (h16*)alloc(1048576);
    float* c1x = alloc(12288); float* c2x = alloc(6144);
    h16* c1f = (h16*)alloc(262144); h16* c2f = (h16*)alloc(262144);
    h16* clsf = (h16*)alloc(4096); h16* h1b = (h16*)alloc(2048); float* logit = alloc(64);
    h16* ttb = (h16*)alloc(2160); h16* ftraj = (h16*)alloc(15360);
    float* fall = alloc(46080); float* hsb = alloc(61440);

    const float *Ws[4], *Gs[4], *Bs[4];
    auto L256 = [&](int total) { return dim3(cdiv(total, 256)); };

    // ================= segmentation SA levels =================
    struct SAcfg { int np, ns; float r2; int base; int ch[4]; const float* in_xyz;
                   const void* in_f; bool f_is_half; int n, Cin; float* out_xyz; h16* out_f; };
    SAcfg sa[4] = {
        {512, 32, 0.01f, 3,  {6, 32, 32, 64},     pcs, pcs, false, 4096, 3,   l1x, l1f},
        {256, 32, 0.04f, 12, {67, 64, 64, 128},   l1x, l1f, true,  512,  64,  l2x, l2f},
        {64,  32, 0.16f, 21, {131, 128, 128, 256},l2x, l2f, true,  256,  128, l3x, l3f},
        {16,  32, 0.64f, 30, {259, 256, 256, 512},l3x, l3f, true,  64,   256, l4x, l4f},
    };
    auto run_sa = [&](const SAcfg& c) {
        int bm = BATCH * c.np, rows = bm * c.ns;
        k_fps<<<BATCH, 256, 0, stream>>>(c.in_xyz, c.n, c.np, fidx);
        k_gather<<<L256(bm * 3), 256, 0, stream>>>(c.in_xyz, fidx, c.out_xyz, c.n, c.np, 3, bm * 3);
        k_ball<<<L256(bm), 256, 0, stream>>>(c.in_xyz, c.out_xyz, c.n, c.np, c.r2, c.ns, gidx, bm);
        if (c.f_is_half)
            k_group<h16><<<L256(rows), 256, 0, stream>>>(c.in_xyz, (const h16*)c.in_f, c.out_xyz,
                                                         gidx, GA, c.n, c.np, c.ns, c.Cin, rows);
        else
            k_group<float><<<L256(rows), 256, 0, stream>>>(c.in_xyz, (const float*)c.in_f, c.out_xyz,
                                                           gidx, GA, c.n, c.np, c.ns, c.Cin, rows);
        mlpP(c.base, 3, Ws, Gs, Bs);
        const h16* gout = run_mlp(GA, GA, GB, GB, Ws, Gs, Bs, c.ch, 3, rows, Wh, stream);
        k_maxpool<<<L256(bm * c.ch[3]), 256, 0, stream>>>(gout, c.out_f, c.ns, c.ch[3], bm * c.ch[3]);
    };
    for (int i = 0; i < 4; ++i) run_sa(sa[i]);

    // ================= feature propagation (i = 3..0) =================
    struct FPcfg { const float* x1; const float* x2; int n1, n2; const void* f1; bool f1h; int C1;
                   const h16* f2; int C2; int base, L; int ch[4]; h16* out; };
    FPcfg fp[4] = {
        {l3x, l4x, 64,  16,  l3f, true,  256, l4f, 512, 60, 2, {768, 256, 256, 0},   n3f},
        {l2x, l3x, 256, 64,  l2f, true,  128, n3f, 256, 54, 2, {384, 256, 256, 0},   n2f},
        {l1x, l2x, 512, 256, l1f, true,  64,  n2f, 256, 48, 2, {320, 256, 128, 0},   n1f},
        {pcs, l1x, 4096,512, pcs, false, 3,   n1f, 128, 39, 3, {131, 128, 128, 128}, l0f},
    };
    for (int i = 0; i < 4; ++i) {
        FPcfg& c = fp[i];
        int bn = BATCH * c.n1, total = bn * (c.C1 + c.C2);
        k_three_nn<<<L256(bn), 256, 0, stream>>>(c.x1, c.x2, c.n1, c.n2, nnid, nnw, bn);
        if (c.f1h)
            k_interp_cat<h16><<<L256(total), 256, 0, stream>>>((const h16*)c.f1, c.f2, nnid, nnw,
                                                               GA, c.n1, c.n2, c.C1, c.C2, total);
        else
            k_interp_cat<float><<<L256(total), 256, 0, stream>>>((const float*)c.f1, c.f2, nnid, nnw,
                                                                 GA, c.n1, c.n2, c.C1, c.C2, total);
        mlpP(c.base, c.L, Ws, Gs, Bs);
        run_mlp(GA, GA, GB, c.out, Ws, Gs, Bs, c.ch, c.L, bn, Wh, stream);
    }

    // whole = relu(bn(l0f @ seg_fc)) ; aff = sigmoid(whole @ affW + affb)
    gemm<h16, h16>(l0f, F(66), F(67), F(68), whole, BATCH * NPTS, 32, 128, 1, Wh, stream);
    k_aff<<<L256(BATCH * NPTS), 256, 0, stream>>>(whole, F(69), F(70), dout + 64, BATCH * NPTS);

    // ================= classifier head =================
    SAcfg cs[2] = {
        {256, 64, 0.04f, 71, {6, 32, 64, 128},    pcs, pcs, false, 4096, 3,   c1x, c1f},
        {128, 64, 0.16f, 80, {131, 128, 128, 256},c1x, c1f, true,  256,  128, c2x, c2f},
    };
    for (int i = 0; i < 2; ++i) run_sa(cs[i]);
    {   // concat(xyz, feats) -> (16*128, 259) -> mlp -> max over 128 -> fc head
        int rows = BATCH * 128, total = rows * 259;
        k_cat_xyzf<<<L256(total), 256, 0, stream>>>(c2x, c2f, GA, 256, total);
        mlpP(89, 3, Ws, Gs, Bs);
        const int ch[4] = {259, 256, 256, 512};
        const h16* gout = run_mlp(GA, GA, GB, GB, Ws, Gs, Bs, ch, 3, rows, Wh, stream);
        k_maxpool<<<L256(BATCH * 512), 256, 0, stream>>>(gout, clsf, 128, 512, BATCH * 512);
        gemm<h16, h16>(clsf, F(98), F(99), F(100), h1b, BATCH, 256, 512, 1, Wh, stream);
        gemm<h16, float>(h1b, F(101), nullptr, F(102), logit, BATCH, 4, 256, 0, Wh, stream);
        k_softmax4<<<1, 32, 0, stream>>>(logit, dout, BATCH);
    }

    // ================= trajectory branch =================
    k_tt<<<L256(BATCH * TSTEPS * 9), 256, 0, stream>>>(traj, pcs, ttb, BATCH * TSTEPS * 9);
    gemm<h16, h16>(ttb, F(103), nullptr, F(104), GA,    BATCH * TSTEPS, 32, 9,  2, Wh, stream);
    gemm<h16, h16>(GA,  F(105), nullptr, F(106), GB,    BATCH * TSTEPS, 64, 32, 2, Wh, stream);
    gemm<h16, h16>(GB,  F(107), nullptr, F(108), ftraj, BATCH * TSTEPS, 64, 64, 0, Wh, stream);
    k_fall<<<L256(BATCH * TSTEPS * 96), 256, 0, stream>>>(whole, ftraj, fall, BATCH * TSTEPS * 96);
    k_lstm<<<BATCH, 128, 0, stream>>>(fall, F(109), F(110), F(111), F(112), hsb);
    gemm<float, float>(hsb, F(113), nullptr, F(114), dout + 64 + BATCH * NPTS,
                       BATCH * TSTEPS, 9, 128, 0, Wh, stream);
}
